// InfoSampling_85667417686374
// MI455X (gfx1250) — matrix operand order repaired and checked by reading, not compiled, not run on hardware
//
#include <hip/hip_runtime.h>
#include <hip/hip_bf16.h>

typedef __attribute__((ext_vector_type(16))) __bf16          v16bf;
typedef __attribute__((ext_vector_type(8)))  float           v8f;
typedef __attribute__((ext_vector_type(16))) unsigned short  us16;

#define KNN 20
#define BN_INV_C 0.9999950000374997f

__device__ __forceinline__ unsigned short f2bfu(float f) {
    union { float f; unsigned u; } v; v.f = f;
    unsigned u = v.u;
    unsigned r = (u + 0x7FFFu + ((u >> 16) & 1u)) >> 16;
    return (unsigned short)r;
}
__device__ __forceinline__ unsigned pack2bf(float lo, float hi) {
    return (unsigned)f2bfu(lo) | ((unsigned)f2bfu(hi) << 16);
}
__device__ __forceinline__ v16bf load_frag(const unsigned short* p) {
    union { us16 u; v16bf b; } t;
    t.u = *(const us16*)p;           // 32B contiguous: 2x b128 loads
    return t.b;
}
__device__ __forceinline__ v8f wmma_bf16(v16bf a, v16bf b, v8f c) {
    return __builtin_amdgcn_wmma_f32_16x16x32_bf16(false, a, false, b, (short)0, c, false, false);
}
__device__ __forceinline__ float gumbelf(unsigned long long idx) {
    unsigned long long z = (idx + 42ULL) * 0x9E3779B97F4A7C15ULL;
    z ^= z >> 30; z *= 0xBF58476D1CE4E5B9ULL;
    z ^= z >> 27; z *= 0x94D049BB133111EBULL;
    z ^= z >> 31;
    float u = ((float)(z >> 40) + 1.0f) * (1.0f / 16777217.0f); // (0,1]
    return -__logf(-__logf(u));
}

// ---------------------------------------------------------------------------
// Pack W (O x C, f32, optionally batched) into WMMA A-fragment order, bf16,
// zero-padded: out[((ot*nk + kc)*32 + lane)*16 + j]
//   lane: lrow = lane&15, kba = (lane&16)?16:0
//   j:    c = kc*32 + kba + j, row = ot*16 + lrow
// ---------------------------------------------------------------------------
__global__ void packW_kernel(const float* __restrict__ W, long wbstride, int O, int C, int nk,
                             unsigned short* __restrict__ out, long obstride) {
    long t = (long)blockIdx.x * blockDim.x + threadIdx.x;   // (ot*nk + kc)*32 + lane
    int b = blockIdx.y;
    long nfrag = (long)(O >> 4) * nk * 32;
    if (t >= nfrag) return;
    int lane = (int)(t & 31);
    long fk = t >> 5;
    int kc = (int)(fk % nk);
    int ot = (int)(fk / nk);
    int lrow = lane & 15;
    int kba = (lane & 16) ? 16 : 0;
    const float* Wb = W + (long)b * wbstride + (long)(ot * 16 + lrow) * C;
    unsigned short* o = out + (long)b * obstride + t * 16;
#pragma unroll
    for (int j = 0; j < 16; ++j) {
        int c = kc * 32 + kba + j;
        o[j] = (c < C) ? f2bfu(Wb[c]) : (unsigned short)0;
    }
}

// ---------------------------------------------------------------------------
// k-NN: top-20 neighbors by -||dp||^2 over the first 3 channels.
// ---------------------------------------------------------------------------
__global__ void knn_kernel(const float* __restrict__ feat, long fbstride, int n_pts,
                           int* __restrict__ idx_out) {
    __shared__ float px[1024], py[1024], pz[1024];
    int b = blockIdx.y;
    const float* f = feat + (long)b * fbstride;
    for (int i = threadIdx.x; i < n_pts; i += blockDim.x) {
        px[i] = f[i]; py[i] = f[n_pts + i]; pz[i] = f[2 * n_pts + i];
    }
    __syncthreads();
    int n = blockIdx.x * blockDim.x + threadIdx.x;
    if (n >= n_pts) return;
    float ax = px[n], ay = py[n], az = pz[n];
    float bv[KNN]; int bi[KNN];
#pragma unroll
    for (int j = 0; j < KNN; ++j) { bv[j] = -3.4e38f; bi[j] = 0; }
    for (int m = 0; m < n_pts; ++m) {
        float dx = ax - px[m], dy = ay - py[m], dz = az - pz[m];
        float d = -(dx * dx + dy * dy + dz * dz);
        if (d > bv[KNN - 1]) {
            int j = KNN - 1;
            while (j > 0 && d > bv[j - 1]) { bv[j] = bv[j - 1]; bi[j] = bi[j - 1]; --j; }
            bv[j] = d; bi[j] = m;
        }
    }
    int* o = idx_out + ((long)b * n_pts + n) * KNN;
#pragma unroll
    for (int j = 0; j < KNN; ++j) o[j] = bi[j];
}

// ---------------------------------------------------------------------------
// Fused EdgeConv with packed weights. One workgroup per (b, n); 4 waves each
// run 16(O)x32(k) WMMA tiles; BN + LeakyReLU + max-over-k epilogue.
// LDS edge tile layout: elds[((kc*32 + col)*32) + klocal] (bf16) so each
// B-fragment is 16 contiguous halves per lane.
// ---------------------------------------------------------------------------
#define EC_THREADS 128
#define EC_WAVES 4
__global__ void edgeconv_kernel(const float* __restrict__ feat, long fbstride, int Cin, int n_pts,
                                const int* __restrict__ knn_idx,
                                const unsigned short* __restrict__ Wp, int nk,
                                const float* __restrict__ g, const float* __restrict__ bias, int O,
                                float* __restrict__ out, long obstride, int ochanoff) {
    __shared__ __align__(32) unsigned short elds[512 * 32];  // [kc][col][klocal]
    __shared__ float tile[EC_WAVES][16][32];
    __shared__ int nbr[32];
    int n = blockIdx.x, b = blockIdx.y;
    int tid = threadIdx.x;
    int C2 = 2 * Cin;
    const float* f = feat + (long)b * fbstride;
    if (tid < 32) nbr[tid] = (tid < KNN) ? knn_idx[((long)b * n_pts + n) * KNN + tid] : n;
    __syncthreads();
    // Stage edge tile (two bf16 per u32 store); unconditional loads + selects.
    unsigned* elds32 = (unsigned*)elds;
    int nu32 = nk * 32 * 16;   // (nk*32 rows) * 32 cols / 2
    for (int idx = tid; idx < nu32; idx += EC_THREADS) {
        int col = idx & 31;
        int k2 = idx >> 5;            // u32 index along k for this col
        int kc = k2 >> 4;
        int kl2 = k2 & 15;
        int k = kc * 32 + kl2 * 2;
        int nb = nbr[col];
        float v0, v1;
        {
            int r = k;
            int ch = (r < Cin) ? r : ((r < C2) ? (r - Cin) : 0);
            float vn = f[(long)ch * n_pts + nb];
            float vc = f[(long)ch * n_pts + n];
            float v = (r < Cin) ? (vn - vc) : vc;
            v0 = (r < C2 && col < KNN) ? v : 0.0f;
        }
        {
            int r = k + 1;
            int ch = (r < Cin) ? r : ((r < C2) ? (r - Cin) : 0);
            float vn = f[(long)ch * n_pts + nb];
            float vc = f[(long)ch * n_pts + n];
            float v = (r < Cin) ? (vn - vc) : vc;
            v1 = (r < C2 && col < KNN) ? v : 0.0f;
        }
        elds32[(kc * 32 + col) * 16 + kl2] = pack2bf(v0, v1);
    }
    __syncthreads();

    int wave = tid >> 5, lane = tid & 31;
    int lrow = lane & 15;
    int kb_b = (lane & 16) ? 16 : 0;
    int iters = (O >> 4) / EC_WAVES;
    for (int it = 0; it < iters; ++it) {
        int ot = it * EC_WAVES + wave;
        v8f acc0 = {}, acc1 = {};
        const unsigned short* wfrag = Wp + ((long)ot * nk * 32 + lane) * 16;
        for (int kc = 0; kc < nk; ++kc) {
            v16bf a  = load_frag(wfrag + (long)kc * 32 * 16);
            v16bf b0 = load_frag(&elds[(kc * 32 + lrow) * 32 + kb_b]);
            v16bf b1 = load_frag(&elds[(kc * 32 + 16 + lrow) * 32 + kb_b]);
            acc0 = wmma_bf16(a, b0, acc0);
            acc1 = wmma_bf16(a, b1, acc1);
        }
        int mbase = (lane & 16) ? 8 : 0;
#pragma unroll
        for (int vv = 0; vv < 8; ++vv) {
            int m = mbase + vv;
            int o = ot * 16 + m;
            float sc = g[o] * BN_INV_C;
            float bb = bias[o];
            float y0 = acc0[vv] * sc + bb;
            float y1 = acc1[vv] * sc + bb;
            y0 = (y0 >= 0.f) ? y0 : 0.2f * y0;
            y1 = (y1 >= 0.f) ? y1 : 0.2f * y1;
            tile[wave][m][lrow] = y0;
            tile[wave][m][16 + lrow] = y1;
        }
        __syncthreads();
        if (tid < EC_WAVES * 16) {
            int w = tid >> 4, m = tid & 15;
            int o = (it * EC_WAVES + w) * 16 + m;
            float mx = -3.4e38f;
#pragma unroll
            for (int k = 0; k < KNN; ++k) mx = fmaxf(mx, tile[w][m][k]);
            out[(long)b * obstride + (long)(ochanoff + o) * n_pts + n] = mx;
        }
        __syncthreads();
    }
}

// ---------------------------------------------------------------------------
// Generic batched WMMA GEMM with packed weights.
// out[b,o,col] = sum_c W[(b),o,c] * X[b,c,col]
// ---------------------------------------------------------------------------
#define GM_THREADS 256
#define GM_WAVES 8
#define GM_CHUNK 128
__global__ void gemm_kernel(const unsigned short* __restrict__ Wp, long wbstride, int nk,
                            const float* __restrict__ X, long xbstride, int C, int ncols,
                            const float* __restrict__ g, const float* __restrict__ bias,
                            int leaky,
                            const float* __restrict__ colmul, long colmul_bstride,
                            float* __restrict__ out, long obstride) {
    __shared__ __align__(32) unsigned short xlds[GM_CHUNK * 32]; // [kc][col][klocal]
    int col0 = blockIdx.x * 32;
    int b = blockIdx.z;
    int tid = threadIdx.x;
    int wave = tid >> 5, lane = tid & 31;
    int ot = blockIdx.y * GM_WAVES + wave;
    int lrow = lane & 15;
    int kb_b = (lane & 16) ? 16 : 0;
    const float* Xb = X + (long)b * xbstride;
    const unsigned short* wbase = Wp + (long)b * wbstride + ((long)ot * nk * 32 + lane) * 16;
    unsigned* xlds32 = (unsigned*)xlds;
    v8f acc0 = {}, acc1 = {};
    for (int cc0 = 0; cc0 < C; cc0 += GM_CHUNK) {
        for (int idx = tid; idx < GM_CHUNK * 16; idx += GM_THREADS) {
            int col = idx & 31;
            int k2 = idx >> 5;
            int kc = k2 >> 4;
            int kl2 = k2 & 15;
            int cc = cc0 + kc * 32 + kl2 * 2;
            int cA = (cc < C) ? cc : (C - 1);
            int cB = (cc + 1 < C) ? (cc + 1) : (C - 1);
            float v0 = Xb[(long)cA * ncols + col0 + col];
            float v1 = Xb[(long)cB * ncols + col0 + col];
            if (cc >= C) v0 = 0.0f;
            if (cc + 1 >= C) v1 = 0.0f;
            xlds32[(kc * 32 + col) * 16 + kl2] = pack2bf(v0, v1);
        }
        __syncthreads();
        int kcg0 = cc0 >> 5;
#pragma unroll
        for (int kc = 0; kc < GM_CHUNK / 32; ++kc) {
            v16bf a  = load_frag(wbase + (long)(kcg0 + kc) * 32 * 16);
            v16bf b0 = load_frag(&xlds[(kc * 32 + lrow) * 32 + kb_b]);
            v16bf b1 = load_frag(&xlds[(kc * 32 + 16 + lrow) * 32 + kb_b]);
            acc0 = wmma_bf16(a, b0, acc0);
            acc1 = wmma_bf16(a, b1, acc1);
        }
        __syncthreads();
    }
    int mbase = (lane & 16) ? 8 : 0;
    int c0 = col0 + lrow, c1 = col0 + 16 + lrow;
    float cm0 = 1.f, cm1 = 1.f;
    if (colmul) { cm0 = colmul[b * colmul_bstride + c0]; cm1 = colmul[b * colmul_bstride + c1]; }
#pragma unroll
    for (int vv = 0; vv < 8; ++vv) {
        int o = ot * 16 + mbase + vv;
        float y0 = acc0[vv], y1 = acc1[vv];
        if (g)         { float sc = g[o] * BN_INV_C; y0 = y0 * sc + bias[o]; y1 = y1 * sc + bias[o]; }
        else if (bias) { y0 += bias[o]; y1 += bias[o]; }
        if (leaky) { y0 = (y0 >= 0.f) ? y0 : 0.2f * y0; y1 = (y1 >= 0.f) ? y1 : 0.2f * y1; }
        y0 *= cm0; y1 *= cm1;
        out[(long)b * obstride + (long)o * ncols + c0] = y0;
        out[(long)b * obstride + (long)o * ncols + c1] = y1;
    }
}

// ---------------------------------------------------------------------------
// hbuf[b, 0:1024, :]   = broadcast(max_n x5[b, ch, :])
// hbuf[b, 1024:1536,:] = xcat[b, ch-1024, :]
// ---------------------------------------------------------------------------
__global__ void assemble_kernel(const float* __restrict__ x5, const float* __restrict__ xcat,
                                float* __restrict__ hbuf, int n_pts) {
    __shared__ float red[256];
    int ch = blockIdx.x, b = blockIdx.y;
    if (ch < 1024) {
        const float* row = x5 + ((long)b * 1024 + ch) * n_pts;
        float mx = -3.4e38f;
        for (int i = threadIdx.x; i < n_pts; i += blockDim.x) mx = fmaxf(mx, row[i]);
        red[threadIdx.x] = mx; __syncthreads();
        for (int s = 128; s > 0; s >>= 1) {
            if (threadIdx.x < s) red[threadIdx.x] = fmaxf(red[threadIdx.x], red[threadIdx.x + s]);
            __syncthreads();
        }
        float m = red[0];
        float* o = hbuf + ((long)b * 1536 + ch) * n_pts;
        for (int i = threadIdx.x; i < n_pts; i += blockDim.x) o[i] = m;
    } else {
        const float* row = xcat + ((long)b * 512 + (ch - 1024)) * n_pts;
        float* o = hbuf + ((long)b * 1536 + ch) * n_pts;
        for (int i = threadIdx.x; i < n_pts; i += blockDim.x) o[i] = row[i];
    }
}

// ---------------------------------------------------------------------------
// Gumbel-softmax over N per (b,s); writes attn (B,S,N) and attn^T (B,N,S).
// ---------------------------------------------------------------------------
__global__ void softmax_kernel(const float* __restrict__ logits, const float* __restrict__ tau,
                               float* __restrict__ attn, float* __restrict__ attnT,
                               int n_pts, int S) {
    __shared__ float red[256];
    __shared__ float zbuf[1024];
    int s = blockIdx.x, b = blockIdx.y;
    float t = tau[0];
    const float* row = logits + ((long)b * S + s) * n_pts;
    float mx = -3.4e38f;
    for (int i = threadIdx.x; i < n_pts; i += blockDim.x) {
        unsigned long long gi = ((unsigned long long)b * S + s) * (unsigned long long)n_pts + i;
        float z = (row[i] + gumbelf(gi)) / t;
        zbuf[i] = z;
        mx = fmaxf(mx, z);
    }
    red[threadIdx.x] = mx; __syncthreads();
    for (int st = 128; st > 0; st >>= 1) {
        if (threadIdx.x < st) red[threadIdx.x] = fmaxf(red[threadIdx.x], red[threadIdx.x + st]);
        __syncthreads();
    }
    mx = red[0]; __syncthreads();
    float sum = 0.f;
    for (int i = threadIdx.x; i < n_pts; i += blockDim.x) {
        float e = __expf(zbuf[i] - mx); zbuf[i] = e; sum += e;
    }
    red[threadIdx.x] = sum; __syncthreads();
    for (int st = 128; st > 0; st >>= 1) {
        if (threadIdx.x < st) red[threadIdx.x] += red[threadIdx.x + st];
        __syncthreads();
    }
    float inv = 1.0f / red[0];
    for (int i = threadIdx.x; i < n_pts; i += blockDim.x) {
        float a = zbuf[i] * inv;
        attn[((long)b * S + s) * n_pts + i] = a;
        attnT[((long)b * n_pts + i) * S + s] = a;
    }
}

// ---------------------------------------------------------------------------
extern "C" void kernel_launch(void* const* d_in, const int* in_sizes, int n_in,
                              void* d_out, int out_size, void* d_ws, size_t ws_size,
                              hipStream_t stream) {
    const float* x     = (const float*)d_in[0];
    const float* prior = (const float*)d_in[1];
    const float* tau   = (const float*)d_in[2];
    const float* W1 = (const float*)d_in[4],  *g1 = (const float*)d_in[5],  *b1 = (const float*)d_in[6];
    const float* W2 = (const float*)d_in[7],  *g2 = (const float*)d_in[8],  *b2 = (const float*)d_in[9];
    const float* W3 = (const float*)d_in[10], *g3 = (const float*)d_in[11], *b3 = (const float*)d_in[12];
    const float* W4 = (const float*)d_in[13], *g4 = (const float*)d_in[14], *b4 = (const float*)d_in[15];
    const float* W5 = (const float*)d_in[16], *g5 = (const float*)d_in[17], *b5 = (const float*)d_in[18];
    const float* W6 = (const float*)d_in[19], *g6 = (const float*)d_in[20], *b6 = (const float*)d_in[21];
    const float* W7 = (const float*)d_in[22], *g7 = (const float*)d_in[23], *b7 = (const float*)d_in[24];
    const float* Wsm = (const float*)d_in[25], *bsv = (const float*)d_in[26];

    const int B = 8, N = 1024, S = 512;
    char* wsp = (char*)d_ws;
    size_t off = 0;
    auto alloc = [&](size_t bytes) -> void* {
        void* p = (void*)(wsp + off);
        off += (bytes + 255) & ~(size_t)255;
        return p;
    };
    int*   knn_idx = (int*)  alloc((size_t)B * N * KNN * sizeof(int));
    float* xcat    = (float*)alloc((size_t)B * 512  * N * sizeof(float));
    float* x5      = (float*)alloc((size_t)B * 1024 * N * sizeof(float));
    float* hbuf    = (float*)alloc((size_t)B * 1536 * N * sizeof(float));
    float* h6      = (float*)alloc((size_t)B * 512  * N * sizeof(float));
    float* h7      = (float*)alloc((size_t)B * 256  * N * sizeof(float));
    float* logits  = (float*)alloc((size_t)B * 512  * N * sizeof(float));
    float* attnT   = (float*)alloc((size_t)B * N * S    * sizeof(float));
    // packed weights (bf16): size = (O/16)*nk*512 halves
    auto palloc = [&](int O, int nk, int nb) -> unsigned short* {
        return (unsigned short*)alloc((size_t)nb * (O / 16) * nk * 512 * sizeof(unsigned short));
    };
    unsigned short* pW1 = palloc(64, 1, 1);    // C2=6   -> nk=1
    unsigned short* pW2 = palloc(64, 4, 1);    // C2=128 -> nk=4
    unsigned short* pW3 = palloc(128, 4, 1);   // C2=128 -> nk=4
    unsigned short* pW4 = palloc(256, 8, 1);   // C2=256 -> nk=8
    unsigned short* pW5 = palloc(1024, 16, 1); // C=512  -> nk=16
    unsigned short* pW6 = palloc(512, 48, 1);  // C=1536 -> nk=48
    unsigned short* pW7 = palloc(256, 16, 1);  // C=512  -> nk=16
    unsigned short* pWs = palloc(512, 8, 1);   // C=256  -> nk=8
    unsigned short* pH7 = palloc(256, 32, B);  // C=1024 -> nk=32, batched

    float* sampled = (float*)d_out;                           // (B,256,S)
    float* attn    = (float*)d_out + (size_t)B * 256 * S;     // (B,S,N)

    auto pack = [&](const float* W, long wbs, int O, int C, int nk, unsigned short* o, long obs, int nb) {
        long nfrag = (long)(O / 16) * nk * 32;
        packW_kernel<<<dim3((unsigned)((nfrag + 255) / 256), nb), dim3(256), 0, stream>>>(
            W, wbs, O, C, nk, o, obs);
    };
    pack(W1, 0, 64, 6, 1, pW1, 0, 1);
    pack(W2, 0, 64, 128, 4, pW2, 0, 1);
    pack(W3, 0, 128, 128, 4, pW3, 0, 1);
    pack(W4, 0, 256, 256, 8, pW4, 0, 1);
    pack(W5, 0, 1024, 512, 16, pW5, 0, 1);
    pack(W6, 0, 512, 1536, 48, pW6, 0, 1);
    pack(W7, 0, 256, 512, 16, pW7, 0, 1);
    pack(Wsm, 0, 512, 256, 8, pWs, 0, 1);

    dim3 kb(128), kg(N / 128, B);
    dim3 eb(EC_THREADS), eg(N, B);

    // Stage 1: x (3ch) -> x1 (64ch @ offset 0)
    knn_kernel<<<kg, kb, 0, stream>>>(x, (long)3 * N, N, knn_idx);
    edgeconv_kernel<<<eg, eb, 0, stream>>>(x, (long)3 * N, 3, N, knn_idx, pW1, 1, g1, b1, 64,
                                           xcat, (long)512 * N, 0);
    // Stage 2: x1 -> x2 (64ch @ offset 64)
    knn_kernel<<<kg, kb, 0, stream>>>(xcat, (long)512 * N, N, knn_idx);
    edgeconv_kernel<<<eg, eb, 0, stream>>>(xcat, (long)512 * N, 64, N, knn_idx, pW2, 4, g2, b2, 64,
                                           xcat, (long)512 * N, 64);
    // Stage 3: x2 -> x3 (128ch @ offset 128)
    knn_kernel<<<kg, kb, 0, stream>>>(xcat + (size_t)64 * N, (long)512 * N, N, knn_idx);
    edgeconv_kernel<<<eg, eb, 0, stream>>>(xcat + (size_t)64 * N, (long)512 * N, 64, N, knn_idx,
                                           pW3, 4, g3, b3, 128, xcat, (long)512 * N, 128);
    // Stage 4: x3 -> x4 (256ch @ offset 256)
    knn_kernel<<<kg, kb, 0, stream>>>(xcat + (size_t)128 * N, (long)512 * N, N, knn_idx);
    edgeconv_kernel<<<eg, eb, 0, stream>>>(xcat + (size_t)128 * N, (long)512 * N, 128, N, knn_idx,
                                           pW4, 8, g4, b4, 256, xcat, (long)512 * N, 256);

    // x5 = block1d(cat, W5)  : (B,1024,N)
    gemm_kernel<<<dim3(N / 32, 1024 / (GM_WAVES * 16), B), dim3(GM_THREADS), 0, stream>>>(
        pW5, 0, 16, xcat, (long)512 * N, 512, N, g5, b5, 1, nullptr, 0, x5, (long)1024 * N);
    // hbuf = [broadcast(max_n x5); xcat]
    assemble_kernel<<<dim3(1536, B), dim3(256), 0, stream>>>(x5, xcat, hbuf, N);
    // h6 = block1d(hbuf, W6) : (B,512,N)
    gemm_kernel<<<dim3(N / 32, 512 / (GM_WAVES * 16), B), dim3(GM_THREADS), 0, stream>>>(
        pW6, 0, 48, hbuf, (long)1536 * N, 1536, N, g6, b6, 1, nullptr, 0, h6, (long)512 * N);
    // h7 = block1d(h6, W7) * prior : (B,256,N)
    gemm_kernel<<<dim3(N / 32, 256 / (GM_WAVES * 16), B), dim3(GM_THREADS), 0, stream>>>(
        pW7, 0, 16, h6, (long)512 * N, 512, N, g7, b7, 1, prior, (long)N, h7, (long)256 * N);
    // logits = Ws @ h7 + bs : (B,512,N)
    gemm_kernel<<<dim3(N / 32, 512 / (GM_WAVES * 16), B), dim3(GM_THREADS), 0, stream>>>(
        pWs, 0, 8, h7, (long)256 * N, 256, N, nullptr, bsv, 0, nullptr, 0, logits, (long)512 * N);
    // attn = softmax((logits + gumbel)/tau) ; also attn^T
    softmax_kernel<<<dim3(S, B), dim3(256), 0, stream>>>(logits, tau, attn, attnT, N, S);
    // pack h7 (batched weights for final contraction)
    pack(h7, (long)256 * N, 256, 1024, 32, pH7, (long)(256 / 16) * 32 * 512, B);
    // sampled[b] = h7[b] (256x1024) @ attnT[b] (1024x512) : (B,256,S)
    gemm_kernel<<<dim3(S / 32, 256 / (GM_WAVES * 16), B), dim3(GM_THREADS), 0, stream>>>(
        pH7, (long)(256 / 16) * 32 * 512, 32, attnT, (long)N * S, N, S,
        nullptr, nullptr, 0, nullptr, 0, sampled, (long)256 * S);
}
